// GeometricGAT_87084756894234
// MI455X (gfx1250) — matrix operand order repaired and checked by reading, not compiled, not run on hardware
//
#include <hip/hip_runtime.h>
#include <math.h>

// ---- types -----------------------------------------------------------------
typedef _Float16 h16;
typedef __attribute__((ext_vector_type(16))) _Float16 v16h;
typedef __attribute__((ext_vector_type(8)))  _Float16 v8h;
typedef __attribute__((ext_vector_type(8)))  float    v8f;

#define DESC 256
#define IN_CH 320
#define HID 256
#define HEADS 4
#define HC 64

// ---- helpers ---------------------------------------------------------------
__device__ __forceinline__ int f32_to_ordkey(float f) {
    int k = __float_as_int(f);
    return k >= 0 ? k : (k ^ 0x7fffffff);
}
__device__ __forceinline__ float ordkey_to_f32(int k) {
    k = k >= 0 ? k : (k ^ 0x7fffffff);
    return __int_as_float(k);
}

// ---- weight / bias prep -----------------------------------------------------
// Builds TRANSPOSED f16 weights:
//   W3t[j][k] = [Wres|Wl|Wr][k][j]   (768 x 320)  -> per-lane B frags contiguous
//   Wpt[j][k] = Wp[k][j]             (256 x 256)
__global__ void gat_prep_weights(const float* __restrict__ Wres, const float* __restrict__ Wl,
                                 const float* __restrict__ Wr,  const float* __restrict__ bres,
                                 const float* __restrict__ bl,  const float* __restrict__ br,
                                 const float* __restrict__ Wp,
                                 h16* __restrict__ W3t, float* __restrict__ b3,
                                 h16* __restrict__ Wpt) {
    int idx = blockIdx.x * blockDim.x + threadIdx.x;
    if (idx < IN_CH * 768) {
        int k = idx / 768, j = idx % 768;
        float v = (j < 256) ? Wres[k * 256 + j]
                : (j < 512) ? Wl[k * 256 + (j - 256)]
                            : Wr[k * 256 + (j - 512)];
        W3t[(size_t)j * IN_CH + k] = (h16)v;            // transposed store
    }
    if (idx < 768)
        b3[idx] = (idx < 256) ? bres[idx] : (idx < 512) ? bl[idx - 256] : br[idx - 512];
    if (idx < 256 * 256) {
        int k = idx / 256, j = idx % 256;
        Wpt[(size_t)j * 256 + k] = (h16)Wp[idx];        // transposed store
    }
}

// ---- per-node positional MLP (3->32 silu ->64) + x row f32->f16 ------------
// one wave (32 lanes) per node
__global__ void gat_pos_feat(const float* __restrict__ kpts, const float* __restrict__ pts3d,
                             const float* __restrict__ W1, const float* __restrict__ b1,
                             const float* __restrict__ W2, const float* __restrict__ b2,
                             const float* __restrict__ x, h16* __restrict__ nodeXh, int N) {
    int n    = blockIdx.x * (blockDim.x >> 5) + (threadIdx.x >> 5);
    int lane = threadIdx.x & 31;
    if (n >= N) return;
    float u = kpts[n * 2 + 0] * (1.0f / 1216.0f);
    float v = kpts[n * 2 + 1] * (1.0f / 352.0f);
    float d = pts3d[n * 3 + 2];
    float hs = u * W1[0 * 32 + lane] + v * W1[1 * 32 + lane] + d * W1[2 * 32 + lane] + b1[lane];
    float hv = hs / (1.0f + expf(-hs));   // silu
    float o0 = b2[lane], o1 = b2[lane + 32];
    #pragma unroll
    for (int h = 0; h < 32; ++h) {
        float hb = __shfl(hv, h, 32);
        o0 += hb * W2[h * 64 + lane];
        o1 += hb * W2[h * 64 + lane + 32];
    }
    h16* dst = nodeXh + (size_t)n * IN_CH;
    dst[DESC + lane]      = (h16)o0;
    dst[DESC + 32 + lane] = (h16)o1;
    const float* xr = x + (size_t)n * DESC;
    #pragma unroll
    for (int i = 0; i < 8; ++i) {
        int c = lane + i * 32;
        dst[c] = (h16)xr[c];
    }
}

// ---- init segment-reduction buffers ---------------------------------------
__global__ void gat_init(int* __restrict__ lmaxb, float* __restrict__ denom,
                         float* __restrict__ accum, int N) {
    int idx = blockIdx.x * blockDim.x + threadIdx.x;
    if (idx < N * HEADS) { lmaxb[idx] = 0x80000000; denom[idx] = 0.0f; }
    if (idx < N * HID) accum[idx] = 0.0f;
}

// ---- WMMA GEMM: C[M x Ncols] = A[M x K](f16) @ Bt[Ncols x K](f16,transposed) + bias
// One wave computes a 16x64 tile: A fragment loaded once per K-chunk, reused
// across 4 v_wmma_f32_16x16x32_f16 issues. All fragment loads are contiguous
// 128-bit vector loads with immediate offsets (K is compile-time, loop fully
// unrolled). No masking needed: a clamped (invalid) A row only affects the
// matching invalid D row, which the guarded store skips.
template <int K>
__global__ void gat_gemm_wmma(const h16* __restrict__ A, const h16* __restrict__ Bt,
                              const float* __restrict__ bias, float* __restrict__ C,
                              int M, int Ncols, int ldc) {
    int tilesN = Ncols >> 6;                       // 64-wide N tiles
    int wave   = blockIdx.x * (blockDim.x >> 5) + (threadIdx.x >> 5);
    int lane   = threadIdx.x & 31;
    int totalT = ((M + 15) >> 4) * tilesN;
    if (wave >= totalT) return;                    // wave-uniform: EXEC all-1 below
    int tm = wave / tilesN, tn = wave % tilesN;

    int  row  = tm * 16 + (lane & 15);
    int  rowc = row < M ? row : (M - 1);           // clamp only (no mask needed)
    int  hi   = lane >> 4;                         // which half-wave: K-half select
    const h16* Arow = A + (size_t)rowc * K + hi * 16;
    int  col0 = tn * 64 + (lane & 15);

    const h16* B0 = Bt + (size_t)(col0     ) * K + hi * 16;
    const h16* B1 = Bt + (size_t)(col0 + 16) * K + hi * 16;
    const h16* B2 = Bt + (size_t)(col0 + 32) * K + hi * 16;
    const h16* B3 = Bt + (size_t)(col0 + 48) * K + hi * 16;

    v8f acc0 = {}, acc1 = {}, acc2 = {}, acc3 = {};
    #pragma unroll
    for (int kb = 0; kb < K; kb += 32) {
        // A 16x32 f16 fragment: per-lane 16 contiguous K halves {kb+off..+15}
        v8h a0 = *(const v8h*)(Arow + kb);
        v8h a1 = *(const v8h*)(Arow + kb + 8);
        v16h a = __builtin_shufflevector(a0, a1, 0, 1, 2, 3, 4, 5, 6, 7,
                                                 8, 9, 10, 11, 12, 13, 14, 15);
        // B 32x16 fragments: per-lane 16 contiguous K halves of column col0+16j
        v16h b0 = *(const v16h*)(B0 + kb);
        v16h b1 = *(const v16h*)(B1 + kb);
        v16h b2 = *(const v16h*)(B2 + kb);
        v16h b3 = *(const v16h*)(B3 + kb);
        acc0 = __builtin_amdgcn_wmma_f32_16x16x32_f16(false, a, false, b0, (short)0, acc0, false, false);
        acc1 = __builtin_amdgcn_wmma_f32_16x16x32_f16(false, a, false, b1, (short)0, acc1, false, false);
        acc2 = __builtin_amdgcn_wmma_f32_16x16x32_f16(false, a, false, b2, (short)0, acc2, false, false);
        acc3 = __builtin_amdgcn_wmma_f32_16x16x32_f16(false, a, false, b3, (short)0, acc3, false, false);
    }
    // C/D 16x16 f32 layout: VGPR r holds row r + 8*hi, column = lane&15
    #pragma unroll
    for (int j = 0; j < 4; ++j) {
        v8f acc = j == 0 ? acc0 : j == 1 ? acc1 : j == 2 ? acc2 : acc3;
        int col = col0 + 16 * j;
        float bv = bias[col];
        #pragma unroll
        for (int r = 0; r < 8; ++r) {
            int m = tm * 16 + r + hi * 8;
            if (m < M) C[(size_t)m * ldc + col] = acc[r] + bv;
        }
    }
}

// ---- edge pass 1: logits + segment max (one wave per edge) -----------------
__global__ void gat_edge_pass1(const int* __restrict__ ei, const float* __restrict__ kpts,
                               const float* __restrict__ C1, const float* __restrict__ We,
                               const float* __restrict__ att, float* __restrict__ elog,
                               int* __restrict__ lmaxb, int E) {
    int e    = blockIdx.x * (blockDim.x >> 5) + (threadIdx.x >> 5);
    int lane = threadIdx.x & 31;
    if (e >= E) return;
    int s = ei[e], d = ei[E + e];
    float rx = (kpts[d * 2 + 0] - kpts[s * 2 + 0]) * (1.0f / 1216.0f);
    float ry = (kpts[d * 2 + 1] - kpts[s * 2 + 1]) * (1.0f / 352.0f);
    float dist = sqrtf(rx * rx + ry * ry);
    const float* xl = C1 + (size_t)s * 768 + 256;        // x_l columns
    const float* xr = C1 + (size_t)d * 768 + 512;        // x_r columns
    float part = 0.0f;
    int ch0 = lane * 8;                                  // 8 lanes per head
    #pragma unroll
    for (int i = 0; i < 8; ++i) {
        int ch = ch0 + i;
        float m = xl[ch] + xr[ch] + rx * We[ch] + ry * We[256 + ch] + dist * We[512 + ch];
        m = m >= 0.0f ? m : 0.2f * m;                    // leaky relu
        part += m * att[ch];                             // att flat == h*64+c
    }
    #pragma unroll
    for (int o = 1; o < 8; o <<= 1) part += __shfl_xor(part, o, 32);
    if ((lane & 7) == 0) {
        int h = lane >> 3;
        elog[(size_t)e * HEADS + h] = part;
        atomicMax(&lmaxb[d * HEADS + h], f32_to_ordkey(part));
    }
}

// ---- edge pass 2: exp + segment sum (one thread per edge*head) -------------
__global__ void gat_edge_pass2(const int* __restrict__ ei, const int* __restrict__ lmaxb,
                               float* __restrict__ elog, float* __restrict__ denom, int E) {
    int idx = blockIdx.x * blockDim.x + threadIdx.x;
    if (idx >= E * HEADS) return;
    int e = idx >> 2, h = idx & 3;
    int d = ei[E + e];
    float ex = expf(elog[idx] - ordkey_to_f32(lmaxb[d * HEADS + h]));
    elog[idx] = ex;
    atomicAdd(&denom[d * HEADS + h], ex);
}

// ---- edge pass 3: alpha out + weighted scatter-add (one wave per edge) -----
__global__ void gat_edge_pass3(const int* __restrict__ ei, const float* __restrict__ C1,
                               const float* __restrict__ elog, const float* __restrict__ denom,
                               float* __restrict__ accum, float* __restrict__ alphaOut, int E) {
    int e    = blockIdx.x * (blockDim.x >> 5) + (threadIdx.x >> 5);
    int lane = threadIdx.x & 31;
    if (e >= E) return;
    int s = ei[e], d = ei[E + e];
    int h = lane >> 3;
    float ex    = elog[(size_t)e * HEADS + h];
    float alpha = ex / (denom[d * HEADS + h] + 1e-16f);
    if ((lane & 7) == 0) alphaOut[(size_t)e * HEADS + h] = alpha;
    const float* xl  = C1 + (size_t)s * 768 + 256;
    float*       acc = accum + (size_t)d * HID;
    int ch0 = lane * 8;
    #pragma unroll
    for (int i = 0; i < 8; ++i) {
        int ch = ch0 + i;
        atomicAdd(&acc[ch], xl[ch] * alpha);
    }
}

// ---- layernorm + silu + residual -> f16 (one wave per node) ----------------
__global__ void gat_finalize(const float* __restrict__ accum, const float* __restrict__ conv_bias,
                             const float* __restrict__ gamma, const float* __restrict__ beta,
                             const float* __restrict__ C1, h16* __restrict__ Yh, int N) {
    int n    = blockIdx.x * (blockDim.x >> 5) + (threadIdx.x >> 5);
    int lane = threadIdx.x & 31;
    if (n >= N) return;
    const float* ar = accum + (size_t)n * HID;
    float v[8], sum = 0.0f, sq = 0.0f;
    #pragma unroll
    for (int i = 0; i < 8; ++i) {
        int ch = lane + i * 32;                          // coalesced
        float t = ar[ch] + conv_bias[ch];
        v[i] = t; sum += t; sq += t * t;
    }
    #pragma unroll
    for (int o = 1; o < 32; o <<= 1) { sum += __shfl_xor(sum, o, 32); sq += __shfl_xor(sq, o, 32); }
    float mu   = sum * (1.0f / 256.0f);
    float var  = sq * (1.0f / 256.0f) - mu * mu;         // biased var, like jnp.var
    float rstd = rsqrtf(var + 1e-5f);
    const float* idr = C1 + (size_t)n * 768;             // identity columns 0..255
    h16* yr = Yh + (size_t)n * HID;
    #pragma unroll
    for (int i = 0; i < 8; ++i) {
        int ch = lane + i * 32;
        float t = (v[i] - mu) * rstd * gamma[ch] + beta[ch];
        t = t / (1.0f + expf(-t));                       // silu
        t += idr[ch];
        yr[ch] = (h16)t;
    }
}

// ---- host launch -----------------------------------------------------------
extern "C" void kernel_launch(void* const* d_in, const int* in_sizes, int n_in,
                              void* d_out, int out_size, void* d_ws, size_t ws_size,
                              hipStream_t stream) {
    const float* x         = (const float*)d_in[0];
    const float* kpts      = (const float*)d_in[1];
    const float* pts3d     = (const float*)d_in[2];
    const int*   ei        = (const int*)  d_in[3];
    const float* W1        = (const float*)d_in[4];
    const float* b1        = (const float*)d_in[5];
    const float* W2        = (const float*)d_in[6];
    const float* b2        = (const float*)d_in[7];
    const float* Wres      = (const float*)d_in[8];
    const float* bres      = (const float*)d_in[9];
    const float* Wl        = (const float*)d_in[10];
    const float* bl        = (const float*)d_in[11];
    const float* Wr        = (const float*)d_in[12];
    const float* br        = (const float*)d_in[13];
    const float* We        = (const float*)d_in[14];
    const float* att       = (const float*)d_in[15];
    const float* conv_bias = (const float*)d_in[16];
    const float* gamma     = (const float*)d_in[17];
    const float* beta      = (const float*)d_in[18];
    const float* Wp        = (const float*)d_in[19];
    const float* bp        = (const float*)d_in[20];

    const int N = in_sizes[0] / DESC;
    const int E = in_sizes[3] / 2;

    float* outMat   = (float*)d_out;                   // N x 256
    float* alphaOut = (float*)d_out + (size_t)N * HID; // E x 4

    // workspace carve-up
    size_t off = 0;
    char*  wsc = (char*)d_ws;
    auto take = [&](size_t bytes) -> void* {
        void* p = wsc + off;
        off += (bytes + 255) & ~(size_t)255;
        return p;
    };
    h16*   nodeXh = (h16*)  take((size_t)N * IN_CH * sizeof(h16));
    h16*   W3t    = (h16*)  take((size_t)768 * IN_CH * sizeof(h16));
    float* b3     = (float*)take(768 * sizeof(float));
    h16*   Wpt    = (h16*)  take((size_t)256 * 256 * sizeof(h16));
    float* C1     = (float*)take((size_t)N * 768 * sizeof(float)); // identity|x_l|x_r
    int*   lmaxb  = (int*)  take((size_t)N * HEADS * sizeof(int));
    float* denom  = (float*)take((size_t)N * HEADS * sizeof(float));
    float* elog   = (float*)take((size_t)E * HEADS * sizeof(float));
    float* accum  = (float*)take((size_t)N * HID * sizeof(float));
    h16*   Yh     = (h16*)  take((size_t)N * HID * sizeof(h16));
    (void)ws_size; (void)n_in; (void)out_size;

    const int TB  = 256;                // 8 waves per block (wave32)
    const int WPB = TB / 32;

    // 1. weights -> transposed f16 (+ fused bias)
    gat_prep_weights<<<(IN_CH * 768 + TB - 1) / TB, TB, 0, stream>>>(
        Wres, Wl, Wr, bres, bl, br, Wp, W3t, b3, Wpt);

    // 2. pos MLP + node_x f16 assembly
    gat_pos_feat<<<(N + WPB - 1) / WPB, TB, 0, stream>>>(
        kpts, pts3d, W1, b1, W2, b2, x, nodeXh, N);

    // 3. init segment buffers
    gat_init<<<((size_t)N * HID + TB - 1) / TB, TB, 0, stream>>>(lmaxb, denom, accum, N);

    // 4. big fused GEMM: node_x(Nx320) @ [Wres|Wl|Wr](320x768) -> C1
    {
        int tiles = ((N + 15) / 16) * (768 / 64);
        gat_gemm_wmma<IN_CH><<<(tiles + WPB - 1) / WPB, TB, 0, stream>>>(
            nodeXh, W3t, b3, C1, N, 768, 768);
    }

    // 5-7. edge softmax + aggregation
    gat_edge_pass1<<<(E + WPB - 1) / WPB, TB, 0, stream>>>(ei, kpts, C1, We, att, elog, lmaxb, E);
    gat_edge_pass2<<<((size_t)E * HEADS + TB - 1) / TB, TB, 0, stream>>>(ei, lmaxb, elog, denom, E);
    gat_edge_pass3<<<(E + WPB - 1) / WPB, TB, 0, stream>>>(ei, C1, elog, denom, accum, alphaOut, E);

    // 8. layernorm + silu + residual -> f16
    gat_finalize<<<(N + WPB - 1) / WPB, TB, 0, stream>>>(accum, conv_bias, gamma, beta, C1, Yh, N);

    // 9. final GEMM: y(Nx256) @ Wp(256x256) + bp -> d_out
    {
        int tiles = ((N + 15) / 16) * (256 / 64);
        gat_gemm_wmma<HID><<<(tiles + WPB - 1) / WPB, TB, 0, stream>>>(
            Yh, Wpt, bp, outMat, N, 256, 256);
    }
}
